// DGN_11931419148972
// MI455X (gfx1250) — matrix operand order repaired and checked by reading, not compiled, not run on hardware
//
#include <hip/hip_runtime.h>
#include <math.h>

#define NN 30000
#define NE 400000
#define FN 64
#define FE 16
#define HID 128
#define OUTF 32
#define LAYERS 4
#define KCAT 2048           // (15+1)*HID
#define KEDGE 288           // 2*HID+FE=272 padded to 9*32

typedef __bf16 bf16_t;
typedef __attribute__((ext_vector_type(16))) __bf16 v16bf;
typedef __attribute__((ext_vector_type(8)))  __bf16 v8bf;
typedef __attribute__((ext_vector_type(8)))  float  v8f;

// ---------- helpers ----------
static __device__ __forceinline__ unsigned short f2bf(float f) {
  unsigned int u = __float_as_uint(f);
  unsigned int r = (u + 0x7FFFu + ((u >> 16) & 1u)) >> 16;
  return (unsigned short)r;
}
static __device__ __forceinline__ int f2ord(float f) {
  int i = __float_as_int(f);
  return i >= 0 ? i : (i ^ 0x7FFFFFFF);
}
static __device__ __forceinline__ float ord2f(int i) {
  int j = i >= 0 ? i : (i ^ 0x7FFFFFFF);
  return __int_as_float(j);
}
static __device__ __forceinline__ v8f vzero8() {
  v8f c; 
#pragma unroll
  for (int i = 0; i < 8; ++i) c[i] = 0.0f;
  return c;
}
// A fragment: 16x32 bf16 tile, row m = lane%16, lanes>=16 hold K 8..15 / 24..31
static __device__ __forceinline__ v16bf frag_a(const unsigned short* row0, int stride, int lane) {
  const unsigned short* p = row0 + (lane & 15) * stride + ((lane & 16) ? 8 : 0);
  union { v16bf v; v8bf h[2]; } u;
  u.h[0] = *reinterpret_cast<const v8bf*>(p);        // K = kb .. kb+7
  u.h[1] = *reinterpret_cast<const v8bf*>(p + 16);   // K = kb+16 .. kb+23
  return u.v;
}
// B fragment: 32x16 bf16 tile; lane holds column n, lanes>=16 hold K 8..15 / 24..31
static __device__ __forceinline__ v16bf frag_b(const unsigned short* wrow_kbase, int lane) {
  const unsigned short* p = wrow_kbase + ((lane & 16) ? 8 : 0);
  union { v16bf v; v8bf h[2]; } u;
  u.h[0] = *reinterpret_cast<const v8bf*>(p);
  u.h[1] = *reinterpret_cast<const v8bf*>(p + 16);
  return u.v;
}
static __device__ __forceinline__ v8f wmma_bf(v16bf a, v16bf b, v8f c) {
  return __builtin_amdgcn_wmma_f32_16x16x32_bf16(false, a, false, b, (short)0, c, false, false);
}
static __device__ __forceinline__ void ln_stats(const float* vbuf, float* smu, float* srs, int tid) {
  if (tid < 16) {
    const float* r = vbuf + tid * HID;
    float m = 0.0f;
    for (int c = 0; c < HID; ++c) m += r[c];
    m *= (1.0f / HID);
    float v = 0.0f;
    for (int c = 0; c < HID; ++c) { float d = r[c] - m; v += d * d; }
    v *= (1.0f / HID);
    smu[tid] = m;
    srs[tid] = rsqrtf(v + 1e-5f);
  }
}

// ---------- prep kernels ----------
__global__ void dgn_cvt(const float* __restrict__ src, unsigned short* __restrict__ dst, int n) {
  int i = blockIdx.x * 256 + threadIdx.x;
  if (i < n) dst[i] = f2bf(src[i]);
}
__global__ void dgn_cvt_mw(const float* __restrict__ src, unsigned short* __restrict__ dst, int total) {
  int i = blockIdx.x * 256 + threadIdx.x;
  if (i >= total) return;
  int k = i % KEDGE;
  int rn = i / KEDGE;
  dst[i] = (k < 2 * HID + FE) ? f2bf(src[rn * (2 * HID + FE) + k]) : (unsigned short)0;
}
__global__ void dgn_prep_nodes(float* deg, float* denom, float* swdx, int n) {
  int i = blockIdx.x * 256 + threadIdx.x;
  if (i < n) { deg[i] = 0.0f; denom[i] = 0.0f; swdx[i] = 0.0f; }
}
__global__ void dgn_deg(const int* __restrict__ src, const int* __restrict__ dst,
                        const float* __restrict__ eig, float* deg, float* denom,
                        float* __restrict__ dwe, int e_count) {
  int e = blockIdx.x * 256 + threadIdx.x;
  if (e >= e_count) return;
  float d = eig[src[e] * 3 + 1] - eig[dst[e] * 3 + 1];
  dwe[e] = d;
  atomicAdd(&deg[dst[e]], 1.0f);
  atomicAdd(&denom[dst[e]], fabsf(d));
}
__global__ void dgn_edgew(const int* __restrict__ dst, const float* __restrict__ dwe,
                          const float* __restrict__ denom, float* __restrict__ wav,
                          float* __restrict__ wdx, float* swdx, int e_count) {
  int e = blockIdx.x * 256 + threadIdx.x;
  if (e >= e_count) return;
  float dn = denom[dst[e]] + 1e-30f;
  float d = dwe[e];
  wav[e] = fabsf(d) / dn;
  float w = d / dn;
  wdx[e] = w;
  atomicAdd(&swdx[dst[e]], w);
}
__global__ void dgn_init_agg(float* s, float* dav, float* dacc, int* mxi, int n) {
  int i = blockIdx.x * 256 + threadIdx.x;
  if (i < n) { s[i] = 0.0f; dav[i] = 0.0f; dacc[i] = 0.0f; mxi[i] = (int)0x80000000; }
}

// ---------- input projection: h = relu(LN(x @ in_w.T + b)) ----------
__global__ __launch_bounds__(256) void dgn_input_proj(
    const float* __restrict__ x, const unsigned short* __restrict__ wbf,
    const float* __restrict__ bias, float* __restrict__ h, unsigned short* __restrict__ hbf) {
  __shared__ unsigned short sa[16 * FN];
  __shared__ float vbuf[16 * HID];
  __shared__ float smu[16], srs[16];
  int tid = threadIdx.x, lane = tid & 31, wv = tid >> 5;
  int n0 = blockIdx.x * 16;
  for (int i = tid; i < 16 * FN; i += 256) {
    int m = i >> 6, k = i & 63;
    sa[i] = f2bf(x[(n0 + m) * FN + k]);
  }
  __syncthreads();
  v8f c = vzero8();
  int nc = wv * 16 + (lane & 15);
#pragma unroll
  for (int kk = 0; kk < FN; kk += 32) {
    v16bf a = frag_a(sa + kk, FN, lane);
    v16bf b = frag_b(wbf + nc * FN + kk, lane);
    c = wmma_bf(a, b, c);
  }
  float bn = bias[nc];
  int rb = (lane & 16) ? 8 : 0;
#pragma unroll
  for (int r = 0; r < 8; ++r) vbuf[(rb + r) * HID + nc] = c[r] + bn;
  __syncthreads();
  ln_stats(vbuf, smu, srs, tid);
  __syncthreads();
  for (int i = tid; i < 16 * HID; i += 256) {
    int m = i >> 7, cch = i & 127;
    float vv = (vbuf[i] - smu[m]) * srs[m];
    vv = vv > 0.0f ? vv : 0.0f;
    h[(n0 + m) * HID + cch] = vv;
    hbf[(n0 + m) * HID + cch] = f2bf(vv);
  }
}

// ---------- edge MLP (WMMA) fused with scatter-aggregate ----------
__global__ __launch_bounds__(256) void dgn_edge(
    const int* __restrict__ srcI, const int* __restrict__ dstI,
    const float* __restrict__ ef, const unsigned short* __restrict__ hbf,
    const unsigned short* __restrict__ mwbf, const float* __restrict__ mb,
    const float* __restrict__ wav, const float* __restrict__ wdx,
    float* __restrict__ s, float* __restrict__ dav, float* __restrict__ dacc,
    int* __restrict__ mxi) {
  __shared__ unsigned short sa[16 * KEDGE];
  __shared__ int sdst[16];
  __shared__ float swa[16], swd[16];
  int tid = threadIdx.x, lane = tid & 31, wv = tid >> 5;
  int e0 = blockIdx.x * 16;
  for (int i = tid; i < 16 * KEDGE; i += 256) {
    int m = i / KEDGE, k = i - m * KEDGE;
    int e = e0 + m;
    unsigned short v;
    if (k < HID)                 v = hbf[(size_t)srcI[e] * HID + k];
    else if (k < 2 * HID)        v = hbf[(size_t)dstI[e] * HID + (k - HID)];
    else if (k < 2 * HID + FE)   v = f2bf(ef[(size_t)e * FE + (k - 2 * HID)]);
    else                         v = 0;
    sa[i] = v;
  }
  if (tid < 16) {
    int e = e0 + tid;
    sdst[tid] = dstI[e];
    swa[tid] = wav[e];
    swd[tid] = wdx[e];
  }
  __syncthreads();
  v8f c = vzero8();
  int nc = wv * 16 + (lane & 15);
  const unsigned short* brow = mwbf + nc * KEDGE;
#pragma unroll
  for (int kk = 0; kk < KEDGE; kk += 32) {
    v16bf a = frag_a(sa + kk, KEDGE, lane);
    v16bf b = frag_b(brow + kk, lane);
    c = wmma_bf(a, b, c);
  }
  float bn = mb[nc];
  int rb = (lane & 16) ? 8 : 0;
#pragma unroll
  for (int r = 0; r < 8; ++r) {
    int m = rb + r;
    float val = c[r] + bn;
    size_t off = (size_t)sdst[m] * HID + nc;
    atomicAdd(s + off, val);
    atomicAdd(dav + off, val * swa[m]);
    atomicAdd(dacc + off, val * swd[m]);
    atomicMax(mxi + off, f2ord(val));
  }
}

// ---------- node update: U GEMM + mixing GEMM + LN + residuals ----------
__global__ __launch_bounds__(256) void dgn_node(
    const float* __restrict__ h,
    const float* __restrict__ s, const float* __restrict__ dav,
    const float* __restrict__ dacc, const int* __restrict__ mxi,
    const float* __restrict__ deg, const float* __restrict__ swdx,
    const unsigned short* __restrict__ uwbf, const float* __restrict__ ub,
    const unsigned short* __restrict__ mixwbf, const float* __restrict__ mixb,
    const float* __restrict__ rs_ptr,
    float* __restrict__ hout, unsigned short* __restrict__ hbf_out) {
  __shared__ unsigned short sx[16 * 1024];   // 32 KB half-stage of concat
  __shared__ unsigned short subf[16 * HID];  // u in bf16
  __shared__ float vbuf[16 * HID];
  __shared__ float sc_deg[16], sc_invd[16], sc_amp[16], sc_att[16], sc_swdx[16];
  __shared__ float smu[16], srs[16];
  int tid = threadIdx.x, lane = tid & 31, wv = tid >> 5;
  int n0 = blockIdx.x * 16;
  if (tid < 16) {
    int n = n0 + tid;
    float dg = deg[n];
    float logd = logf(dg + 1.0f);
    sc_deg[tid] = dg;
    sc_invd[tid] = 1.0f / fmaxf(dg, 1.0f);
    sc_amp[tid] = logd;                                        // DELTA=1
    sc_att[tid] = (dg > 0.0f) ? (1.0f / (logd > 0.0f ? logd : 1.0f)) : 0.0f;
    sc_swdx[tid] = swdx[n];
  }
  __syncthreads();
  v8f c = vzero8();
  int nc = wv * 16 + (lane & 15);
  const unsigned short* brow = uwbf + nc * KCAT;
  for (int half = 0; half < 2; ++half) {
    for (int i = tid; i < 16 * 1024; i += 256) {
      int m = i >> 10, kk = i & 1023;
      int k = half * 1024 + kk;
      int n = n0 + m;
      float val;
      if (k < HID) {
        val = h[(size_t)n * HID + k];
      } else {
        int j = k - HID;
        int blk = j / 640;                 // 0=identity 1=amp 2=att
        int a = (j - blk * 640) >> 7;      // aggregator
        int cch = j & 127;
        size_t off = (size_t)n * HID + cch;
        float base;
        if (a == 0)      base = s[off] * sc_invd[m];                       // mean
        else if (a == 1) base = (sc_deg[m] > 0.0f) ? ord2f(mxi[off]) : 0.0f; // max
        else if (a == 2) base = s[off];                                    // sum
        else if (a == 3) base = dav[off];                                  // dir_av
        else             base = fabsf(dacc[off] - sc_swdx[m] * h[off]);    // dir_dx
        float scal = (blk == 0) ? 1.0f : ((blk == 1) ? sc_amp[m] : sc_att[m]);
        val = base * scal;
      }
      sx[i] = f2bf(val);
    }
    __syncthreads();
#pragma unroll 8
    for (int kk = 0; kk < 1024; kk += 32) {
      v16bf a = frag_a(sx + kk, 1024, lane);
      v16bf b = frag_b(brow + half * 1024 + kk, lane);
      c = wmma_bf(a, b, c);
    }
    __syncthreads();
  }
  const float GSN = rsqrtf((float)NN) * rsqrtf(1.0f + 1e-5f);  // graph-size + BN(eval)
  float bn = ub[nc];
  int rb = (lane & 16) ? 8 : 0;
#pragma unroll
  for (int r = 0; r < 8; ++r) subf[(rb + r) * HID + nc] = f2bf((c[r] + bn) * GSN);
  __syncthreads();
  // mixing GEMM: [16x128] @ [128x128]
  v8f c2 = vzero8();
  const unsigned short* brow2 = mixwbf + nc * HID;
#pragma unroll
  for (int kk = 0; kk < HID; kk += 32) {
    v16bf a = frag_a(subf + kk, HID, lane);
    v16bf b = frag_b(brow2 + kk, lane);
    c2 = wmma_bf(a, b, c2);
  }
  float bn2 = mixb[nc];
#pragma unroll
  for (int r = 0; r < 8; ++r) {
    int m = rb + r;
    float v = c2[r] + bn2;
    v = v > 0.0f ? v : 0.01f * v;           // leaky_relu
    v += h[(size_t)(n0 + m) * HID + nc];    // internal PNA residual
    vbuf[m * HID + nc] = v;
  }
  __syncthreads();
  ln_stats(vbuf, smu, srs, tid);
  __syncthreads();
  float sig = 1.0f / (1.0f + expf(-rs_ptr[0]));
  for (int i = tid; i < 16 * HID; i += 256) {
    int m = i >> 7, cch = i & 127;
    float xv = (vbuf[i] - smu[m]) * srs[m];
    xv = xv > 0.0f ? xv : 0.0f;
    float hn = xv + sig * h[(size_t)(n0 + m) * HID + cch];
    hout[(size_t)(n0 + m) * HID + cch] = hn;
    hbf_out[(size_t)(n0 + m) * HID + cch] = f2bf(hn);
  }
}

// ---------- output head ----------
__global__ __launch_bounds__(256) void dgn_out(
    const unsigned short* __restrict__ hbf,
    const unsigned short* __restrict__ w1, const float* __restrict__ b1,
    const unsigned short* __restrict__ w2, const float* __restrict__ b2,
    float* __restrict__ out) {
  __shared__ float vbuf[16 * HID];
  __shared__ unsigned short ybf[16 * HID];
  __shared__ float smu[16], srs[16];
  int tid = threadIdx.x, lane = tid & 31, wv = tid >> 5;
  int n0 = blockIdx.x * 16;
  v8f c = vzero8();
  int nc = wv * 16 + (lane & 15);
  const unsigned short* arow0 = hbf + (size_t)n0 * HID;
  const unsigned short* brow = w1 + nc * HID;
#pragma unroll
  for (int kk = 0; kk < HID; kk += 32) {
    v16bf a = frag_a(arow0 + kk, HID, lane);
    v16bf b = frag_b(brow + kk, lane);
    c = wmma_bf(a, b, c);
  }
  float bn = b1[nc];
  int rb = (lane & 16) ? 8 : 0;
#pragma unroll
  for (int r = 0; r < 8; ++r) vbuf[(rb + r) * HID + nc] = c[r] + bn;
  __syncthreads();
  ln_stats(vbuf, smu, srs, tid);
  __syncthreads();
  for (int i = tid; i < 16 * HID; i += 256) {
    int m = i >> 7;
    float xv = (vbuf[i] - smu[m]) * srs[m];
    ybf[i] = f2bf(xv > 0.0f ? xv : 0.0f);
  }
  __syncthreads();
  if (wv < 2) {
    v8f c2 = vzero8();
    int nc2 = wv * 16 + (lane & 15);
    const unsigned short* brow2 = w2 + nc2 * HID;
#pragma unroll
    for (int kk = 0; kk < HID; kk += 32) {
      v16bf a = frag_a(ybf + kk, HID, lane);
      v16bf b = frag_b(brow2 + kk, lane);
      c2 = wmma_bf(a, b, c2);
    }
    float bn2 = b2[nc2];
#pragma unroll
    for (int r = 0; r < 8; ++r)
      out[(size_t)(n0 + rb + r) * OUTF + nc2] = c2[r] + bn2;
  }
}

// ---------- host ----------
extern "C" void kernel_launch(void* const* d_in, const int* in_sizes, int n_in,
                              void* d_out, int out_size, void* d_ws, size_t ws_size,
                              hipStream_t stream) {
  const float* x        = (const float*)d_in[0];
  const float* ef       = (const float*)d_in[1];
  const int*   srcI     = (const int*)d_in[2];
  const int*   dstI     = (const int*)d_in[3];
  const float* eig      = (const float*)d_in[4];
  const float* in_w     = (const float*)d_in[5];
  const float* in_b     = (const float*)d_in[6];
  const float* M_w      = (const float*)d_in[7];
  const float* M_b      = (const float*)d_in[8];
  const float* U_w      = (const float*)d_in[9];
  const float* U_b      = (const float*)d_in[10];
  const float* mix_w    = (const float*)d_in[11];
  const float* mix_b    = (const float*)d_in[12];
  const float* res_sc   = (const float*)d_in[13];
  const float* o1w      = (const float*)d_in[14];
  const float* o1b      = (const float*)d_in[15];
  const float* o2w      = (const float*)d_in[16];
  const float* o2b      = (const float*)d_in[17];

  size_t off = 0;
  auto alloc = [&](size_t bytes) -> void* {
    void* p = (char*)d_ws + off;
    off += (bytes + 255) & ~(size_t)255;
    return p;
  };
  float* h0   = (float*)alloc((size_t)NN * HID * 4);
  float* h1   = (float*)alloc((size_t)NN * HID * 4);
  unsigned short* hbf = (unsigned short*)alloc((size_t)NN * HID * 2);
  float* sAgg = (float*)alloc((size_t)NN * HID * 4);
  float* dav  = (float*)alloc((size_t)NN * HID * 4);
  float* dacc = (float*)alloc((size_t)NN * HID * 4);
  int*   mxi  = (int*)alloc((size_t)NN * HID * 4);
  float* deg   = (float*)alloc((size_t)NN * 4);
  float* denom = (float*)alloc((size_t)NN * 4);
  float* swdx  = (float*)alloc((size_t)NN * 4);
  float* dwe = (float*)alloc((size_t)NE * 4);
  float* wav = (float*)alloc((size_t)NE * 4);
  float* wdx = (float*)alloc((size_t)NE * 4);
  unsigned short* inwb  = (unsigned short*)alloc((size_t)HID * FN * 2);
  unsigned short* mwb   = (unsigned short*)alloc((size_t)LAYERS * HID * KEDGE * 2);
  unsigned short* uwb   = (unsigned short*)alloc((size_t)LAYERS * HID * KCAT * 2);
  unsigned short* mixwb = (unsigned short*)alloc((size_t)LAYERS * HID * HID * 2);
  unsigned short* o1wb  = (unsigned short*)alloc((size_t)HID * HID * 2);
  unsigned short* o2wb  = (unsigned short*)alloc((size_t)OUTF * HID * 2);

  auto cdiv = [](int a, int b) { return (a + b - 1) / b; };

  dgn_cvt<<<cdiv(HID * FN, 256), 256, 0, stream>>>(in_w, inwb, HID * FN);
  dgn_cvt_mw<<<cdiv(LAYERS * HID * KEDGE, 256), 256, 0, stream>>>(M_w, mwb, LAYERS * HID * KEDGE);
  dgn_cvt<<<cdiv(LAYERS * HID * KCAT, 256), 256, 0, stream>>>(U_w, uwb, LAYERS * HID * KCAT);
  dgn_cvt<<<cdiv(LAYERS * HID * HID, 256), 256, 0, stream>>>(mix_w, mixwb, LAYERS * HID * HID);
  dgn_cvt<<<cdiv(HID * HID, 256), 256, 0, stream>>>(o1w, o1wb, HID * HID);
  dgn_cvt<<<cdiv(OUTF * HID, 256), 256, 0, stream>>>(o2w, o2wb, OUTF * HID);

  dgn_prep_nodes<<<cdiv(NN, 256), 256, 0, stream>>>(deg, denom, swdx, NN);
  dgn_deg<<<cdiv(NE, 256), 256, 0, stream>>>(srcI, dstI, eig, deg, denom, dwe, NE);
  dgn_edgew<<<cdiv(NE, 256), 256, 0, stream>>>(dstI, dwe, denom, wav, wdx, swdx, NE);

  dgn_input_proj<<<NN / 16, 256, 0, stream>>>(x, inwb, in_b, h0, hbf);

  float* hc = h0;
  float* hn = h1;
  for (int l = 0; l < LAYERS; ++l) {
    dgn_init_agg<<<cdiv(NN * HID, 256), 256, 0, stream>>>(sAgg, dav, dacc, mxi, NN * HID);
    dgn_edge<<<NE / 16, 256, 0, stream>>>(srcI, dstI, ef, hbf,
                                          mwb + (size_t)l * HID * KEDGE, M_b + l * HID,
                                          wav, wdx, sAgg, dav, dacc, mxi);
    dgn_node<<<NN / 16, 256, 0, stream>>>(hc, sAgg, dav, dacc, mxi, deg, swdx,
                                          uwb + (size_t)l * HID * KCAT, U_b + l * HID,
                                          mixwb + (size_t)l * HID * HID, mix_b + l * HID,
                                          res_sc + l, hn, hbf);
    float* t = hc; hc = hn; hn = t;
  }

  dgn_out<<<NN / 16, 256, 0, stream>>>(hbf, o1wb, o1b, o2wb, o2b, (float*)d_out);
}
